// MoEClassifier_67199058313278
// MI455X (gfx1250) — matrix operand order repaired and checked by reading, not compiled, not run on hardware
//
#include <hip/hip_runtime.h>
#include <math.h>

// ---------------- dims ----------------
constexpr int B_  = 16, S_ = 512, H_ = 768, NH_ = 8, HD_ = 96;
constexpr int FF_ = 3072, E_ = 8, FE_ = 3072, C_ = 1000, L_ = 6;
constexpr long BS_ = (long)B_ * S_;            // 8192 rows

#define DEVINL __device__ __forceinline__

typedef __attribute__((ext_vector_type(16))) __bf16 bf16x16;
typedef __attribute__((ext_vector_type(8)))  float  f32x8;

union FragU { bf16x16 v; uint4 u[2]; };

DEVINL unsigned short f2bf(float f) {
  union { float f; unsigned int u; } c; c.f = f;
  unsigned int u = c.u;
  u += 0x7fffu + ((u >> 16) & 1u);            // round-to-nearest-even
  return (unsigned short)(u >> 16);
}
DEVINL float bf2f(unsigned short h) {
  union { unsigned int u; float f; } c; c.u = ((unsigned int)h) << 16;
  return c.f;
}
DEVINL float gelu_f(float x) { return 0.5f * x * (1.0f + erff(x * 0.7071067811865476f)); }

DEVINL f32x8 wmma_bf16(bf16x16 a, bf16x16 b, f32x8 c) {
  // v_wmma_f32_16x16x32_bf16: D = A(16x32) * B(32x16) + C(16x16 f32)
  return __builtin_amdgcn_wmma_f32_16x16x32_bf16(false, a, false, b, (short)0, c, false, false);
}
DEVINL f32x8 zero8() { f32x8 z = {0.f,0.f,0.f,0.f,0.f,0.f,0.f,0.f}; return z; }

// A-fragment (16x32 bf16), LDS row-major tile, 'stride' ushorts per row.
// ISA layout: lane m=L%16, half=L/16; VGPR v: k = (v/4)*16 + half*8 + (v%4)*2
DEVINL bf16x16 load_frag_a(const unsigned short* p0, int stride, int lane) {
  int m = lane & 15, half = lane >> 4;
  const unsigned short* p = p0 + m * stride + half * 8;
  FragU f;
  f.u[0] = *(const uint4*)(p);        // k = half*8 .. half*8+7
  f.u[1] = *(const uint4*)(p + 16);   // k = 16+half*8 .. +7
  return f.v;
}
// B-fragment (32x16 bf16) from LDS transposed tile Bt[n][k], 'stride' per n-row.
// ISA layout: lane n=L%16, half=L/16; VGPR v: k = (v/4)*16 + half*8 + (v%4)*2
DEVINL bf16x16 load_frag_b(const unsigned short* p0, int stride, int lane) {
  int n = lane & 15, half = lane >> 4;
  const unsigned short* p = p0 + n * stride + half * 8;
  FragU f;
  f.u[0] = *(const uint4*)(p);        // k = half*8 .. half*8+7
  f.u[1] = *(const uint4*)(p + 16);   // k = 16+half*8 .. +7
  return f.v;
}

// ---------------- embedding ----------------
__global__ __launch_bounds__(256) void embed_kernel(
    const int* __restrict__ ids, const float* __restrict__ tok,
    const float* __restrict__ pos, float* __restrict__ x)
{
  long i = (long)blockIdx.x * 256 + threadIdx.x;
  if (i >= BS_ * H_) return;
  int h = (int)(i % H_);
  long bs = i / H_;
  int s = (int)(bs % S_);
  int id = ids[bs];
  x[i] = tok[(long)id * H_ + h] + pos[(long)s * H_ + h];
}

// ---------------- layernorm (H=768, one row per block) ----------------
template<bool OUTF>
__global__ __launch_bounds__(256) void layernorm_kernel(
    const float* __restrict__ x, const float* __restrict__ g, const float* __restrict__ beta,
    unsigned short* __restrict__ outB, float* __restrict__ outF)
{
  int row = blockIdx.x;
  const float* xr = x + (long)row * H_;
  float v[3]; float s = 0.f, s2 = 0.f;
#pragma unroll
  for (int j = 0; j < 3; ++j) { v[j] = xr[threadIdx.x + j * 256]; s += v[j]; s2 += v[j] * v[j]; }
#pragma unroll
  for (int off = 16; off; off >>= 1) { s += __shfl_xor(s, off); s2 += __shfl_xor(s2, off); }
  __shared__ float rs[8], rs2[8];
  int lane = threadIdx.x & 31, wave = threadIdx.x >> 5;
  if (lane == 0) { rs[wave] = s; rs2[wave] = s2; }
  __syncthreads();
  if (threadIdx.x == 0) {
    float a = 0.f, b = 0.f;
    for (int i = 0; i < 8; ++i) { a += rs[i]; b += rs2[i]; }
    float mean = a * (1.f / H_);
    float var  = b * (1.f / H_) - mean * mean;
    rs[0] = mean; rs2[0] = rsqrtf(var + 1e-5f);
  }
  __syncthreads();
  float mean = rs[0], rstd = rs2[0];
#pragma unroll
  for (int j = 0; j < 3; ++j) {
    int i = threadIdx.x + j * 256;
    float o = (v[j] - mean) * rstd * g[i] + beta[i];
    outB[(long)row * H_ + i] = f2bf(o);
    if (OUTF) outF[(long)row * H_ + i] = o;
  }
}

// ---------------- generic bf16 WMMA GEMM (double-buffered LDS) ----------------
// C[M,N] = epi(A_bf16[M,K] @ W_f32->bf16[K,N] + bias)
// Block tile 128x64, BK=32; 4 waves, each wave 32 rows x 64 cols
//   = 2 A-frags x 4 B-frags = 8 WMMAs per K-step (B-frag reused across M-subtiles).
// ACT: 0 none, 1 exact GELU.  EPI: 0 store f32, 1 res+store f32, 2 store bf16
// z-batched over W/bias/out via strides (A shared across z).
template<int ACT, int EPI>
__global__ __launch_bounds__(128) void gemm_kernel(
    const unsigned short* __restrict__ A, const float* __restrict__ W,
    const float* __restrict__ bias, const float* __restrict__ res,
    float* __restrict__ outF, unsigned short* __restrict__ outB,
    int M, int N, int K, long wStrideZ, long bStrideZ, long oStrideZ)
{
  constexpr int BM = 128, BN = 64, BK = 32, PAD = 8, LDW = BK + PAD;
  __shared__ __align__(16) unsigned short As[2][BM][LDW];   // 2 x 128x40
  __shared__ __align__(16) unsigned short Bt[2][BN][LDW];   // 2 x 64x40 (transposed)
  int z = blockIdx.z;
  const float* Wz = W + (long)z * wStrideZ;
  const float* bz = bias + (long)z * bStrideZ;
  int n0 = blockIdx.x * BN, m0 = blockIdx.y * BM;
  int tid = threadIdx.x, lane = tid & 31, wave = tid >> 5;

  auto stage = [&](int buf, int k0) {
    // A: 128x32 bf16, 8-halfword vector chunks: 512 chunks / 128 thr = 4 each
    for (int i = tid; i < BM * 4; i += 128) {
      int r = i >> 2, c8 = (i & 3) * 8;
      int gr = m0 + r;
      uint4 val = make_uint4(0u, 0u, 0u, 0u);
      if (gr < M) val = *(const uint4*)(A + (long)gr * K + k0 + c8);
      *(uint4*)&As[buf][r][c8] = val;
    }
    // W: 32x64 f32 -> bf16, transposed into Bt[n][k]: 512 float4 / 128 thr = 4 each
    for (int i = tid; i < BK * 16; i += 128) {
      int r = i >> 4, c4 = (i & 15) * 4;
      float4 wv = *(const float4*)(Wz + (long)(k0 + r) * N + n0 + c4);
      Bt[buf][c4 + 0][r] = f2bf(wv.x); Bt[buf][c4 + 1][r] = f2bf(wv.y);
      Bt[buf][c4 + 2][r] = f2bf(wv.z); Bt[buf][c4 + 3][r] = f2bf(wv.w);
    }
  };

  f32x8 acc[2][4] = { { zero8(), zero8(), zero8(), zero8() },
                      { zero8(), zero8(), zero8(), zero8() } };
  int nk = K / BK;
  stage(0, 0);
  __syncthreads();
  for (int kt = 0; kt < nk; ++kt) {
    int cur = kt & 1;
    if (kt + 1 < nk) {
      if (kt + 2 < nk) {                        // global_prefetch_b8 two tiles ahead
        __builtin_prefetch(Wz + (long)((kt + 2) * BK + (tid & 31)) * N + n0, 0, 1);
      }
      stage(cur ^ 1, (kt + 1) * BK);            // fill other buffer while computing
    }
    bf16x16 a0 = load_frag_a(&As[cur][wave * 32][0], LDW, lane);
    bf16x16 a1 = load_frag_a(&As[cur][wave * 32 + 16][0], LDW, lane);
#pragma unroll
    for (int t = 0; t < 4; ++t) {
      bf16x16 bfr = load_frag_b(&Bt[cur][t * 16][0], LDW, lane);
      acc[0][t] = wmma_bf16(a0, bfr, acc[0][t]);
      acc[1][t] = wmma_bf16(a1, bfr, acc[1][t]);
    }
    __syncthreads();
  }

  // epilogue (C layout: VGPR r -> row r + 8*(lane/16); col = lane%16)
  int half = lane >> 4, c = lane & 15;
#pragma unroll
  for (int g = 0; g < 2; ++g) {
#pragma unroll
    for (int t = 0; t < 4; ++t) {
      int gc = n0 + t * 16 + c;
      float bv = bz[gc];
#pragma unroll
      for (int r = 0; r < 8; ++r) {
        int gr = m0 + wave * 32 + g * 16 + r + half * 8;
        if (gr < M) {
          float v = acc[g][t][r] + bv;
          if (ACT == 1) v = gelu_f(v);
          long idx = (long)z * oStrideZ + (long)gr * N + gc;
          if (EPI == 0)      outF[idx] = v;
          else if (EPI == 1) outF[idx] = res[idx] + v;
          else               outB[idx] = f2bf(v);
        }
      }
    }
  }
}

// ---------------- fused attention ----------------
// grid: b(16) x head(8) x qtile(16 of 32 rows); block = 64 threads (2 waves)
__global__ __launch_bounds__(64) void attention_kernel(
    const float* __restrict__ qkv,            // [B*S, 2304]
    const int* __restrict__ amask,            // [B, S]
    unsigned short* __restrict__ obf)         // [B*S, 768] bf16
{
  __shared__ __align__(16) unsigned short Qs[32][104];
  __shared__ __align__(16) unsigned short Kt[16][104];
  __shared__ __align__(16) unsigned short Sc[32][520];  // scores -> probs (bf16)
  __shared__ __align__(16) unsigned short Vt[96][40];
  __shared__ float mbias[512];
  __shared__ float rowSum[32];
  __shared__ float red2[32][2];

  int qt = blockIdx.x & 15;
  int head = (blockIdx.x >> 4) & 7;
  int b = blockIdx.x >> 7;
  int q0 = qt * 32;
  int tid = threadIdx.x, lane = tid & 31, wave = tid >> 5;
  const float scale = 0.1020620726159658f;    // 1/sqrt(96)

  for (int i = tid; i < 512; i += 64)
    mbias[i] = (amask[b * 512 + i] > 0) ? 0.f : -1e9f;
  for (int i = tid; i < 32 * 96; i += 64) {
    int r = i / 96, d = i % 96;
    Qs[r][d] = f2bf(qkv[(long)(b * 512 + q0 + r) * 2304 + head * 96 + d]);
  }
  __syncthreads();

  // --- scores = scale * Q K^T + mask ---
  for (int kt = 0; kt < 32; ++kt) {
    for (int i = tid; i < 16 * 96; i += 64) {
      int r = i / 96, d = i % 96;
      Kt[r][d] = f2bf(qkv[(long)(b * 512 + kt * 16 + r) * 2304 + 768 + head * 96 + d]);
    }
    __syncthreads();
    f32x8 sacc = zero8();
#pragma unroll
    for (int kc = 0; kc < 3; ++kc) {
      bf16x16 qa = load_frag_a(&Qs[wave * 16][kc * 32], 104, lane);
      bf16x16 kb = load_frag_b(&Kt[0][kc * 32], 104, lane);
      sacc = wmma_bf16(qa, kb, sacc);
    }
    int half = lane >> 4, c = lane & 15;
#pragma unroll
    for (int r = 0; r < 8; ++r) {
      int lr = wave * 16 + r + half * 8;
      int key = kt * 16 + c;
      Sc[lr][key] = f2bf(sacc[r] * scale + mbias[key]);
    }
    __syncthreads();
  }

  // --- softmax (unnormalized exp; 1/sum folded into PV epilogue) ---
  int srow = tid >> 1, part = tid & 1;
  float mx = -1e30f;
  for (int cc = part * 256; cc < part * 256 + 256; ++cc)
    mx = fmaxf(mx, bf2f(Sc[srow][cc]));
  red2[srow][part] = mx;
  __syncthreads();
  float rmax = fmaxf(red2[srow][0], red2[srow][1]);
  __syncthreads();
  float sm = 0.f;
  for (int cc = part * 256; cc < part * 256 + 256; ++cc) {
    float p = __expf(bf2f(Sc[srow][cc]) - rmax);
    Sc[srow][cc] = f2bf(p);
    sm += p;
  }
  red2[srow][part] = sm;
  __syncthreads();
  if (part == 0) rowSum[srow] = red2[srow][0] + red2[srow][1];
  __syncthreads();

  // --- O = P V ---
  f32x8 oacc[6] = { zero8(), zero8(), zero8(), zero8(), zero8(), zero8() };
  for (int ch = 0; ch < 16; ++ch) {
    for (int i = tid; i < 96 * 32; i += 64) {
      int d = i >> 5, kk = i & 31;
      Vt[d][kk] = f2bf(qkv[(long)(b * 512 + ch * 32 + kk) * 2304 + 1536 + head * 96 + d]);
    }
    __syncthreads();
    bf16x16 pa = load_frag_a(&Sc[wave * 16][ch * 32], 520, lane);
#pragma unroll
    for (int t = 0; t < 6; ++t) {
      bf16x16 vb = load_frag_b(&Vt[t * 16][0], 40, lane);
      oacc[t] = wmma_bf16(pa, vb, oacc[t]);
    }
    __syncthreads();
  }
  int half = lane >> 4, c = lane & 15;
#pragma unroll
  for (int t = 0; t < 6; ++t)
#pragma unroll
    for (int r = 0; r < 8; ++r) {
      int lr = wave * 16 + r + half * 8;
      float inv = 1.f / rowSum[lr];
      obf[(long)(b * 512 + q0 + lr) * 768 + head * 96 + t * 16 + c] = f2bf(oacc[t][r] * inv);
    }
}

// ---------------- masked mean pool ----------------
__global__ __launch_bounds__(256) void pool_kernel(
    const float* __restrict__ xf, const int* __restrict__ amask,
    float* __restrict__ pooled, unsigned short* __restrict__ pooledbf)
{
  int b = blockIdx.x;
  float ms = 0.f;
  for (int s = threadIdx.x; s < 512; s += 256) ms += (amask[b * 512 + s] > 0) ? 1.f : 0.f;
#pragma unroll
  for (int off = 16; off; off >>= 1) ms += __shfl_xor(ms, off);
  __shared__ float mred[8];
  __shared__ float msum_s;
  if ((threadIdx.x & 31) == 0) mred[threadIdx.x >> 5] = ms;
  __syncthreads();
  if (threadIdx.x == 0) {
    float t = 0.f; for (int i = 0; i < 8; ++i) t += mred[i];
    msum_s = fmaxf(t, 1.f);
  }
  __syncthreads();
  float inv = 1.f / msum_s;
  for (int h = threadIdx.x; h < 768; h += 256) {
    float acc = 0.f;
    for (int s = 0; s < 512; ++s) {
      float m = (amask[b * 512 + s] > 0) ? 1.f : 0.f;
      acc += xf[((long)b * 512 + s) * 768 + h] * m;
    }
    acc *= inv;
    pooled[b * 768 + h] = acc;
    pooledbf[b * 768 + h] = f2bf(acc);
  }
}

// ---------------- router gate (16x8 softmax) ----------------
__global__ __launch_bounds__(128) void gate_kernel(
    const float* __restrict__ pooled, const float* __restrict__ Wr,
    const float* __restrict__ br, float* __restrict__ gate)
{
  int t = threadIdx.x;               // exactly 128 = 16*8
  int b = t >> 3, e = t & 7;
  float acc = br[e];
  for (int h = 0; h < 768; ++h) acc += pooled[b * 768 + h] * Wr[h * 8 + e];
  __shared__ float lg[16][8];
  lg[b][e] = acc;
  __syncthreads();
  float mx = lg[b][0];
#pragma unroll
  for (int i = 1; i < 8; ++i) mx = fmaxf(mx, lg[b][i]);
  float sum = 0.f;
#pragma unroll
  for (int i = 0; i < 8; ++i) sum += __expf(lg[b][i] - mx);
  gate[t] = __expf(acc - mx) / sum;
}

// ---------------- MoE expert FFN2 + gated combine ----------------
// out[16,1000] = sum_e gate[:,e] * (eh[e] @ We2[e] + be2[e]);  M=16 = one WMMA tile
__global__ __launch_bounds__(128) void moe_out_kernel(
    const unsigned short* __restrict__ ehbf,   // [E][16][3072] bf16
    const float* __restrict__ We2,             // [E][3072][1000]
    const float* __restrict__ be2,             // [E][1000]
    const float* __restrict__ gate,            // [16][8]
    float* __restrict__ outp)                  // [16][1000]
{
  __shared__ __align__(16) unsigned short As[16][40];
  __shared__ __align__(16) unsigned short Bt[64][40];
  __shared__ float gsh[16][8];
  int n0 = blockIdx.x * 64;
  int tid = threadIdx.x, lane = tid & 31, wave = tid >> 5;
  gsh[tid >> 3][tid & 7] = gate[tid];
  f32x8 outAcc = zero8();
  int half = lane >> 4, c = lane & 15;
  int gc = n0 + wave * 16 + c;

  for (int e = 0; e < E_; ++e) {
    f32x8 acc = zero8();
    for (int k0 = 0; k0 < FE_; k0 += 32) {
      for (int i = tid; i < 16 * 4; i += 128) {
        int r = i >> 2, c8 = (i & 3) * 8;
        *(uint4*)&As[r][c8] = *(const uint4*)(ehbf + ((long)e * 16 + r) * FE_ + k0 + c8);
      }
      for (int i = tid; i < 32 * 16; i += 128) {
        int r = i >> 4, c4 = (i & 15) * 4;
#pragma unroll
        for (int j = 0; j < 4; ++j) {
          int col = n0 + c4 + j;
          float wv = (col < C_) ? We2[((long)e * FE_ + k0 + r) * C_ + col] : 0.f;
          Bt[c4 + j][r] = f2bf(wv);
        }
      }
      __syncthreads();
      bf16x16 af = load_frag_a(&As[0][0], 40, lane);
      bf16x16 bfr = load_frag_b(&Bt[wave * 16][0], 40, lane);
      acc = wmma_bf16(af, bfr, acc);
      __syncthreads();
    }
    float bv = (gc < C_) ? be2[e * C_ + gc] : 0.f;
#pragma unroll
    for (int r = 0; r < 8; ++r) {
      int row = r + half * 8;
      outAcc[r] += gsh[row][e] * (acc[r] + bv);
    }
  }
  if (gc < C_) {
#pragma unroll
    for (int r = 0; r < 8; ++r)
      outp[(long)(r + half * 8) * C_ + gc] = outAcc[r];
  }
}

// ---------------- host launcher ----------------
extern "C" void kernel_launch(void* const* d_in, const int* in_sizes, int n_in,
                              void* d_out, int out_size, void* d_ws, size_t ws_size,
                              hipStream_t stream) {
  (void)in_sizes; (void)n_in; (void)out_size; (void)ws_size;
  const int*   input_ids = (const int*)d_in[0];
  const int*   amask     = (const int*)d_in[1];
  const float* tok       = (const float*)d_in[2];
  const float* pos       = (const float*)d_in[3];
  const float* Wqkv      = (const float*)d_in[4];
  const float* bqkv      = (const float*)d_in[5];
  const float* Wo        = (const float*)d_in[6];
  const float* bo        = (const float*)d_in[7];
  const float* ln1g      = (const float*)d_in[8];
  const float* ln1b      = (const float*)d_in[9];
  const float* ln2g      = (const float*)d_in[10];
  const float* ln2b      = (const float*)d_in[11];
  const float* W1        = (const float*)d_in[12];
  const float* b1        = (const float*)d_in[13];
  const float* W2        = (const float*)d_in[14];
  const float* b2        = (const float*)d_in[15];
  const float* lnfg      = (const float*)d_in[16];
  const float* lnfb      = (const float*)d_in[17];
  const float* Wr        = (const float*)d_in[18];
  const float* br        = (const float*)d_in[19];
  const float* We1       = (const float*)d_in[20];
  const float* be1       = (const float*)d_in[21];
  const float* We2       = (const float*)d_in[22];
  const float* be2       = (const float*)d_in[23];
  float* out = (float*)d_out;

  // workspace carve-up (256B aligned slices)
  char* w = (char*)d_ws;
  auto take = [&](long bytes) { char* p = w; w += (bytes + 255) & ~255L; return p; };
  float*          x        = (float*)take(BS_ * H_ * 4);            // 25.2 MB
  float*          qkv      = (float*)take(BS_ * 3 * H_ * 4);        // 75.5 MB
  unsigned short* hbf      = (unsigned short*)take(BS_ * H_ * 2);   // 12.6 MB
  unsigned short* obf      = (unsigned short*)take(BS_ * H_ * 2);   // 12.6 MB
  unsigned short* ffbf     = (unsigned short*)take(BS_ * FF_ * 2);  // 50.3 MB
  float*          pooled   = (float*)take(B_ * H_ * 4);
  unsigned short* pooledbf = (unsigned short*)take(B_ * H_ * 2);
  float*          gate     = (float*)take(B_ * E_ * 4);
  unsigned short* ehbf     = (unsigned short*)take((long)E_ * B_ * FE_ * 2);
  float*          xf       = qkv;   // reuse qkv region for final LN output

  // embeddings
  embed_kernel<<<(int)((BS_ * H_ + 255) / 256), 256, 0, stream>>>(input_ids, tok, pos, x);

  for (int l = 0; l < L_; ++l) {
    const float* Wqkv_l = Wqkv + (long)l * H_ * 3 * H_;
    const float* bqkv_l = bqkv + (long)l * 3 * H_;
    const float* Wo_l   = Wo   + (long)l * H_ * H_;
    const float* bo_l   = bo   + (long)l * H_;
    const float* W1_l   = W1   + (long)l * H_ * FF_;
    const float* b1_l   = b1   + (long)l * FF_;
    const float* W2_l   = W2   + (long)l * FF_ * H_;
    const float* b2_l   = b2   + (long)l * H_;

    layernorm_kernel<false><<<(int)BS_, 256, 0, stream>>>(
        x, ln1g + l * H_, ln1b + l * H_, hbf, nullptr);
    gemm_kernel<0, 0><<<dim3(3 * H_ / 64, (int)(BS_ / 128), 1), 128, 0, stream>>>(
        hbf, Wqkv_l, bqkv_l, nullptr, qkv, nullptr, (int)BS_, 3 * H_, H_, 0, 0, 0);
    attention_kernel<<<B_ * NH_ * (S_ / 32), 64, 0, stream>>>(qkv, amask, obf);
    gemm_kernel<0, 1><<<dim3(H_ / 64, (int)(BS_ / 128), 1), 128, 0, stream>>>(
        obf, Wo_l, bo_l, x, x, nullptr, (int)BS_, H_, H_, 0, 0, 0);
    layernorm_kernel<false><<<(int)BS_, 256, 0, stream>>>(
        x, ln2g + l * H_, ln2b + l * H_, hbf, nullptr);
    gemm_kernel<1, 2><<<dim3(FF_ / 64, (int)(BS_ / 128), 1), 128, 0, stream>>>(
        hbf, W1_l, b1_l, nullptr, nullptr, ffbf, (int)BS_, FF_, H_, 0, 0, 0);
    gemm_kernel<0, 1><<<dim3(H_ / 64, (int)(BS_ / 128), 1), 128, 0, stream>>>(
        ffbf, W2_l, b2_l, x, x, nullptr, (int)BS_, H_, FF_, 0, 0, 0);
  }

  // final LN (f32 for pooling), pool, gate
  layernorm_kernel<true><<<(int)BS_, 256, 0, stream>>>(x, lnfg, lnfb, hbf, xf);
  pool_kernel<<<B_, 256, 0, stream>>>(xf, amask, pooled, pooledbf);
  gate_kernel<<<1, 128, 0, stream>>>(pooled, Wr, br, gate);

  // expert FFN1 (z-batched over 8 experts), GELU -> bf16
  gemm_kernel<1, 2><<<dim3(FE_ / 64, 1, E_), 128, 0, stream>>>(
      pooledbf, We1, be1, nullptr, nullptr, ehbf, B_, FE_, H_,
      (long)H_ * FE_, (long)FE_, (long)B_ * FE_);

  // expert FFN2 + gated combine -> out [16, 1000]
  moe_out_kernel<<<(C_ + 63) / 64, 128, 0, stream>>>(ehbf, We2, be2, gate, out);
}
